// HeteroGAT_35175782154437
// MI455X (gfx1250) — matrix operand order repaired and checked by reading, not compiled, not run on hardware
//
#include <hip/hip_runtime.h>
#include <stdint.h>

#define HEADS 6
#define HID 64
#define HC 384        // HEADS*HID
#define IN_DIM 768
#define BATCH 64
#define POOLW 1920    // 5*HC
#define L1IN 2688     // 5*HC + IN_DIM

typedef __attribute__((ext_vector_type(16))) __bf16 v16bf;
typedef __attribute__((ext_vector_type(8)))  float  v8f;

// ---------- float <-> order-preserving uint (for atomic max on float) ----------
__device__ __forceinline__ unsigned f2ord(float f) {
  unsigned u = __float_as_uint(f);
  return (u & 0x80000000u) ? ~u : (u | 0x80000000u);
}
__device__ __forceinline__ float ord2f(unsigned u) {
  unsigned v = (u & 0x80000000u) ? (u & 0x7fffffffu) : ~u;
  return __uint_as_float(v);
}

// ---------- weight convert + transpose: W[768,384] f32 -> Wt[384,768] bf16 ----------
__global__ __launch_bounds__(256) void wcvt_kernel(const float* __restrict__ W,
                                                   __bf16* __restrict__ Wt) {
  int i = blockIdx.x * 256 + threadIdx.x;       // over HC*IN_DIM
  if (i >= HC * IN_DIM) return;
  int n = i / IN_DIM, k = i - n * IN_DIM;
  Wt[i] = (__bf16)W[(size_t)k * HC + n];
}

// ---------- tiled BF16 WMMA GEMM: C[M,384] = A[M,768] @ W + bias ----------
// Wt is [384,768] (output-channel major, K contiguous).
// One wave -> 16x64 output tile (4 accumulators): the A fragment is loaded and
// converted once per k-step and reused by 4 independent WMMAs (A traffic /4,
// and the 4 independent WMMAs cover the bf16 WMMA result latency).
// Block = 4 waves covering 4 consecutive M tiles; grid = (ceil(M/64), 6).
__global__ __launch_bounds__(128) void gemm_bf16_wmma(
    const float* __restrict__ A, int M,
    const __bf16* __restrict__ Wt,
    const float* __restrict__ bias,
    float* __restrict__ C) {
  int wave  = threadIdx.x >> 5;
  int lane  = threadIdx.x & 31;
  int m0    = (blockIdx.x * 4 + wave) * 16;
  int n0    = blockIdx.y * 64;                  // 4 consecutive 16-wide N tiles
  if (m0 >= M) return;                          // uniform per wave
  int group = lane >> 4;                        // 0 or 1
  int lm    = lane & 15;
  int arow  = (m0 + lm) < M ? (m0 + lm) : (M - 1);   // clamp OOB rows for loads
  const float*  ap  = A  + (size_t)arow * IN_DIM;
  const __bf16* bp0 = Wt + (size_t)(n0 +  0 + lm) * IN_DIM;
  const __bf16* bp1 = Wt + (size_t)(n0 + 16 + lm) * IN_DIM;
  const __bf16* bp2 = Wt + (size_t)(n0 + 32 + lm) * IN_DIM;
  const __bf16* bp3 = Wt + (size_t)(n0 + 48 + lm) * IN_DIM;

  v8f acc0 = {}, acc1 = {}, acc2 = {}, acc3 = {};
  for (int k0 = 0; k0 < IN_DIM; k0 += 32) {
    // A fragment (16x32, contiguous K order shared with B): lane m = lane&15, element e -> K = e + group*16
    v16bf a;
#pragma unroll
    for (int e = 0; e < 16; ++e) {
      int k = k0 + e + group * 16;
      a[e] = (__bf16)ap[k];
    }
    // B fragments: lane n, element e -> K = e + group*16 (K-contiguous in Wt)
    v16bf b0, b1, b2, b3;
#pragma unroll
    for (int e = 0; e < 16; ++e) {
      int k = k0 + e + group * 16;
      b0[e] = bp0[k]; b1[e] = bp1[k]; b2[e] = bp2[k]; b3[e] = bp3[k];
    }
    acc0 = __builtin_amdgcn_wmma_f32_16x16x32_bf16(false, a, false, b0, (short)0, acc0, false, false);
    acc1 = __builtin_amdgcn_wmma_f32_16x16x32_bf16(false, a, false, b1, (short)0, acc1, false, false);
    acc2 = __builtin_amdgcn_wmma_f32_16x16x32_bf16(false, a, false, b2, (short)0, acc2, false, false);
    acc3 = __builtin_amdgcn_wmma_f32_16x16x32_bf16(false, a, false, b3, (short)0, acc3, false, false);
  }
  float bv0 = bias[n0 + lm];
  float bv1 = bias[n0 + 16 + lm];
  float bv2 = bias[n0 + 32 + lm];
  float bv3 = bias[n0 + 48 + lm];
#pragma unroll
  for (int r = 0; r < 8; ++r) {                 // D VGPR r holds row m = r + 8*group
    int m = m0 + r + 8 * group;
    if (m < M) {
      float* crow = C + (size_t)m * HC + n0 + lm;
      crow[0]  = acc0[r] + bv0;
      crow[16] = acc1[r] + bv1;
      crow[32] = acc2[r] + bv2;
      crow[48] = acc3[r] + bv3;
    }
  }
}

// ---------- edge pass A: logits per (edge,head) + segmented max ----------
__global__ __launch_bounds__(256) void edge_logits_max(
    const float* __restrict__ xl, const float* __restrict__ xr,
    const float* __restrict__ att, const int* __restrict__ edge, int E,
    float* __restrict__ logits, unsigned* __restrict__ mbuf) {
  int lane = threadIdx.x & 31;
  int e = blockIdx.x * 8 + (threadIdx.x >> 5);
  if (e >= E) return;
  int src = edge[e], dst = edge[E + e];
  const float* pl = xl + (size_t)src * HC;
  const float* pr = xr + (size_t)dst * HC;
#pragma unroll
  for (int h = 0; h < HEADS; ++h) {
    int c1 = h * HID + lane, c2 = c1 + 32;
    float v1 = pl[c1] + pr[c1]; v1 = v1 > 0.f ? v1 : 0.2f * v1;  // leaky 0.2
    float v2 = pl[c2] + pr[c2]; v2 = v2 > 0.f ? v2 : 0.2f * v2;
    float v = v1 * att[c1] + v2 * att[c2];
#pragma unroll
    for (int off = 16; off > 0; off >>= 1) v += __shfl_xor(v, off, 32);
    if (lane == 0) {
      logits[(size_t)e * HEADS + h] = v;
      atomicMax(&mbuf[(size_t)dst * HEADS + h], f2ord(v));
    }
  }
}

// ---------- edge pass B: a = exp(logit - m), denom += a ----------
__global__ __launch_bounds__(256) void edge_exp_denom(
    float* __restrict__ logits, const unsigned* __restrict__ mbuf,
    const int* __restrict__ edge, int E, float* __restrict__ denom) {
  int i = blockIdx.x * 256 + threadIdx.x;
  if (i >= E * HEADS) return;
  int e = i / HEADS, h = i - e * HEADS;
  int dst = edge[E + e];
  float m = ord2f(mbuf[(size_t)dst * HEADS + h]);
  unsigned mu = __float_as_uint(m);
  if ((mu & 0x7f800000u) == 0x7f800000u) m = 0.f;   // empty segment -> 0 (matches ref)
  float a = __expf(logits[i] - m);
  logits[i] = a;
  atomicAdd(&denom[(size_t)dst * HEADS + h], a);
}

// ---------- edge pass C: acc[dst] += alpha * xl[src] ----------
__global__ __launch_bounds__(256) void edge_scatter(
    const float* __restrict__ xl, const float* __restrict__ logits,
    const float* __restrict__ denom, const int* __restrict__ edge, int E,
    float* __restrict__ acc) {
  int lane = threadIdx.x & 31;
  int e = blockIdx.x * 8 + (threadIdx.x >> 5);
  if (e >= E) return;
  int src = edge[e], dst = edge[E + e];
  const float* pl = xl + (size_t)src * HC;
  float* pd = acc + (size_t)dst * HC;
#pragma unroll
  for (int h = 0; h < HEADS; ++h) {
    float alpha = logits[(size_t)e * HEADS + h] /
                  (denom[(size_t)dst * HEADS + h] + 1e-16f);
    int c1 = h * HID + lane, c2 = c1 + 32;
    atomicAdd(&pd[c1], alpha * pl[c1]);
    atomicAdd(&pd[c2], alpha * pl[c2]);
  }
}

// ---------- bias sum + leaky(0.01) + mean-pool accumulate ----------
__global__ __launch_bounds__(256) void finalize_pool(
    const float* __restrict__ acc, const float* __restrict__ b0,
    const float* __restrict__ b1, const float* __restrict__ b2,
    const int* __restrict__ batch, int n, int coff, int tIdx,
    float* __restrict__ pooled, float* __restrict__ cnt) {
  int i = blockIdx.x * 256 + threadIdx.x;
  if (i >= n * HC) return;
  int node = i / HC, c = i - node * HC;
  float v = acc[i] + b0[c];
  if (b1) v += b1[c];
  if (b2) v += b2[c];
  v = v > 0.f ? v : 0.01f * v;
  int b = batch[node];
  atomicAdd(&pooled[(size_t)b * L1IN + coff + c], v);
  if (c == 0) atomicAdd(&cnt[b * 5 + tIdx], 1.0f);
}

__global__ __launch_bounds__(256) void pool_divide(float* __restrict__ pooled,
                                                   const float* __restrict__ cnt) {
  int i = blockIdx.x * 256 + threadIdx.x;
  if (i >= BATCH * POOLW) return;
  int b = i / POOLW, r = i - b * POOLW;
  int t = r / HC;
  pooled[(size_t)b * L1IN + r] /= fmaxf(cnt[b * 5 + t], 1.0f);
}

__global__ __launch_bounds__(256) void copy_post(const float* __restrict__ post,
                                                 float* __restrict__ pooled) {
  int i = blockIdx.x * 256 + threadIdx.x;
  if (i >= BATCH * IN_DIM) return;
  int b = i / IN_DIM, k = i - b * IN_DIM;
  pooled[(size_t)b * L1IN + POOLW + k] = post[i];
}

// ---------- final MLP 2688->64->2 + softmax ----------
__global__ __launch_bounds__(64) void mlp_head(
    const float* __restrict__ concat, const float* __restrict__ W1,
    const float* __restrict__ b1, const float* __restrict__ W2,
    const float* __restrict__ b2, float* __restrict__ out) {
  __shared__ float h[64];
  __shared__ float o[2];
  int b = blockIdx.x, j = threadIdx.x;
  const float* row = concat + (size_t)b * L1IN;
  float a = b1[j];
  for (int k = 0; k < L1IN; ++k) a += row[k] * W1[(size_t)k * 64 + j];
  h[j] = a > 0.f ? a : 0.01f * a;
  __syncthreads();
  if (j < 2) {
    float s = b2[j];
    for (int t = 0; t < 64; ++t) s += h[t] * W2[t * 2 + j];
    o[j] = s > 0.f ? s : 0.01f * s;
  }
  __syncthreads();
  if (j < 2) {
    float m  = fmaxf(o[0], o[1]);
    float e0 = __expf(o[0] - m), e1 = __expf(o[1] - m);
    out[b * 2 + j] = (j == 0 ? e0 : e1) / (e0 + e1);
  }
}

extern "C" void kernel_launch(void* const* d_in, const int* in_sizes, int n_in,
                              void* d_out, int out_size, void* d_ws, size_t ws_size,
                              hipStream_t stream) {
  (void)n_in; (void)out_size; (void)ws_size;
  // ---- input map (setup_inputs dict order, params dict nested-flattened) ----
  const float* x[5]; int nt[5];
  for (int t = 0; t < 5; ++t) { x[t] = (const float*)d_in[t]; nt[t] = in_sizes[t] / IN_DIM; }
  const float* post = (const float*)d_in[5];
  auto P = [&](int r, int j) { return (const float*)d_in[6 + r * 6 + j]; }; // 0:Wl 1:bl 2:Wr 3:br 4:att 5:bias
  const float* W1  = (const float*)d_in[66];
  const float* b1v = (const float*)d_in[67];
  const float* W2  = (const float*)d_in[68];
  const float* b2v = (const float*)d_in[69];
  const int* edges[10]; int Ec[10]; int maxE = 0;
  for (int r = 0; r < 10; ++r) {
    edges[r] = (const int*)d_in[70 + r];
    Ec[r] = in_sizes[70 + r] / 2;
    if (Ec[r] > maxE) maxE = Ec[r];
  }
  const int* batch[5];
  for (int t = 0; t < 5; ++t) batch[t] = (const int*)d_in[80 + t];

  static const int rs[10] = {0, 0, 0, 1, 1, 2, 3, 4, 2, 3};
  static const int rd[10] = {2, 3, 4, 2, 3, 0, 0, 0, 1, 1};

  int maxN = 0, totalN = 0, accOff[5];
  for (int t = 0; t < 5; ++t) { accOff[t] = totalN; totalN += nt[t]; if (nt[t] > maxN) maxN = nt[t]; }

  // ---- workspace bump allocator ----
  char* p = (char*)d_ws;
  auto alloc = [&](size_t bytes) -> char* {
    char* q = p; p += (bytes + 255) & ~(size_t)255; return q;
  };
  __bf16*   wt     = (__bf16*)  alloc((size_t)20 * HC * IN_DIM * sizeof(__bf16));
  float*    xl     = (float*)   alloc((size_t)maxN * HC * sizeof(float));
  float*    xr     = (float*)   alloc((size_t)maxN * HC * sizeof(float));
  float*    lg     = (float*)   alloc((size_t)maxE * HEADS * sizeof(float));
  unsigned* mb     = (unsigned*)alloc((size_t)maxN * HEADS * sizeof(unsigned));
  float*    dn     = (float*)   alloc((size_t)maxN * HEADS * sizeof(float));
  float*    acc    = (float*)   alloc((size_t)totalN * HC * sizeof(float));
  float*    pooled = (float*)   alloc((size_t)BATCH * L1IN * sizeof(float));
  float*    cnt    = (float*)   alloc((size_t)BATCH * 5 * sizeof(float));

  hipMemsetAsync(acc,    0, (size_t)totalN * HC * sizeof(float), stream);
  hipMemsetAsync(pooled, 0, (size_t)BATCH * L1IN * sizeof(float), stream);
  hipMemsetAsync(cnt,    0, (size_t)BATCH * 5 * sizeof(float), stream);

  // ---- weight convert + transpose (bf16, K-contiguous) ----
  int wthr = (HC * IN_DIM + 255) / 256;
  for (int r = 0; r < 10; ++r) {
    wcvt_kernel<<<wthr, 256, 0, stream>>>(P(r, 0), wt + (size_t)(2 * r + 0) * HC * IN_DIM);
    wcvt_kernel<<<wthr, 256, 0, stream>>>(P(r, 2), wt + (size_t)(2 * r + 1) * HC * IN_DIM);
  }

  // ---- per-relation: projections (WMMA GEMM) + GATv2 edge softmax/aggregate ----
  for (int r = 0; r < 10; ++r) {
    int s = rs[r], d = rd[r];
    int Ms = nt[s], Md = nt[d], E = Ec[r];
    gemm_bf16_wmma<<<dim3((Ms + 63) / 64, 6), 128, 0, stream>>>(
        x[s], Ms, wt + (size_t)(2 * r + 0) * HC * IN_DIM, P(r, 1), xl);
    gemm_bf16_wmma<<<dim3((Md + 63) / 64, 6), 128, 0, stream>>>(
        x[d], Md, wt + (size_t)(2 * r + 1) * HC * IN_DIM, P(r, 3), xr);
    hipMemsetAsync(mb, 0, (size_t)Md * HEADS * sizeof(unsigned), stream); // f2ord-min
    hipMemsetAsync(dn, 0, (size_t)Md * HEADS * sizeof(float), stream);
    edge_logits_max<<<(E + 7) / 8, 256, 0, stream>>>(xl, xr, P(r, 4), edges[r], E, lg, mb);
    edge_exp_denom<<<(E * HEADS + 255) / 256, 256, 0, stream>>>(lg, mb, edges[r], E, dn);
    edge_scatter<<<(E + 7) / 8, 256, 0, stream>>>(xl, lg, dn, edges[r], E,
                                                  acc + (size_t)accOff[d] * HC);
  }

  // ---- per-type: bias sums, leaky(0.01), mean pool into concat buffer ----
  const float* tb[5][3] = {
      {P(5, 5), P(6, 5), P(7, 5)},   // tag    <- q->tag, a->tag, c->tag
      {P(8, 5), P(9, 5), nullptr},   // module <- q->mod, a->mod
      {P(0, 5), P(3, 5), nullptr},   // question <- tag->q, mod->q
      {P(1, 5), P(4, 5), nullptr},   // answer   <- tag->a, mod->a
      {P(2, 5), nullptr, nullptr},   // comment  <- tag->c
  };
  for (int t = 0; t < 5; ++t) {
    finalize_pool<<<(nt[t] * HC + 255) / 256, 256, 0, stream>>>(
        acc + (size_t)accOff[t] * HC, tb[t][0], tb[t][1], tb[t][2],
        batch[t], nt[t], t * HC, t, pooled, cnt);
  }
  pool_divide<<<(BATCH * POOLW + 255) / 256, 256, 0, stream>>>(pooled, cnt);
  copy_post<<<(BATCH * IN_DIM + 255) / 256, 256, 0, stream>>>(post, pooled);

  // ---- MLP head + softmax ----
  mlp_head<<<BATCH, 64, 0, stream>>>(pooled, W1, b1v, W2, b2v, (float*)d_out);
}
